// GlobalAttn_58746562675126
// MI455X (gfx1250) — matrix operand order repaired and checked by reading, not compiled, not run on hardware
//
#include <hip/hip_runtime.h>
#include <hip/hip_bf16.h>

// GlobalAttn for MI455X (gfx1250): 5-stage WMMA pipeline, f16 intermediates.
// B=4 C=128 H=W=360 r=9 -> h=w=40, P=1600, r2=81 (padded to 96 for 16x16 tiles).
// Memory-bound (~71 GFLOP vs >=1 GB traffic @ 23.3 TB/s): f16 intermediates,
// 'a' projection folded into G = Wa^T K, and all WMMA fragments loaded as
// contiguous 128-bit LDS vectors (B tiles stored transposed [n][k]).

typedef _Float16 v16h __attribute__((ext_vector_type(16)));
typedef _Float16 v8h  __attribute__((ext_vector_type(8)));
typedef _Float16 v4h  __attribute__((ext_vector_type(4)));
typedef float    v8f  __attribute__((ext_vector_type(8)));

#define BATCH 4
#define CCH   128
#define HH    360
#define WW    360
#define RR    9
#define R2    81
#define R2P   96
#define HSUB  40
#define NPOS  1600   /* 40*40 */
#define NBC   512    /* BATCH*CCH */
#define HWPIX 129600 /* 360*360 */

static __device__ inline v8f wmma_f16(v16h a, v16h b, v8f c) {
  // D = A(16x32 f16) * B(32x16 f16) + C(16x16 f32)
  return __builtin_amdgcn_wmma_f32_16x16x32_f16(false, a, false, b,
                                                (short)0, c, false, false);
}

static __device__ inline v16h cat8(v8h lo, v8h hi) {
  return __builtin_shufflevector(lo, hi, 0, 1, 2, 3, 4, 5, 6, 7,
                                 8, 9, 10, 11, 12, 13, 14, 15);
}

// A-matrix fragment (16x32 f16) from row-major [m][k], leading dim ldk (halfs).
// Per lane: e=0..7 -> k=kbase..kbase+7, e=8..15 -> k=kbase+16..+23:
// two contiguous 16B chunks -> 2x ds_load_b128. (kbase = kOff + 8*(lane>=16))
static __device__ inline v16h load_A_lds(const _Float16* base, int m, int ldk, int kOff) {
  const int lane = threadIdx.x & 31;
  const int kbase = kOff + ((lane >> 4) << 3);
  const _Float16* p = base + m * ldk + kbase;
  return cat8(*(const v8h*)p, *(const v8h*)(p + 16));
}

// B-matrix fragment (32x16 f16) from TRANSPOSED storage [n][k], ld ldk (halfs).
// Per lane: e=0..7 -> k=kbase..kbase+7, e=8..15 -> k=kbase+16..+23 -> 2x ds_load_b128.
// (kbase = kOff + 8*(lane>=16))
static __device__ inline v16h load_B_ldsT(const _Float16* base, int n, int ldk, int kOff) {
  const int lane = threadIdx.x & 31;
  const int kbase = kOff + ((lane >> 4) << 3);
  const _Float16* p = base + n * ldk + kbase;
  return cat8(*(const v8h*)p, *(const v8h*)(p + 16));
}

// ---------------------------------------------------------------------------
// Kernel 1: 1x1 conv as GEMM; scatter-store f16 into rearranged X2[bc][j][p].
// grid (2025, 4), block 256 (8 waves). Each wave: Mtile=wave, 4 Ntiles of 16.
// ---------------------------------------------------------------------------
__global__ __launch_bounds__(256) void k1_conv(
    const float* __restrict__ x, const float* __restrict__ Wconv,
    const float* __restrict__ bconv, _Float16* __restrict__ X2) {
  __shared__ __align__(16) _Float16 Ws[CCH * CCH];  // [o][c] row-major
  __shared__ __align__(16) _Float16 XsT[64 * CCH];  // [n][c] transposed
  const int tid = threadIdx.x;
  const int b = blockIdx.y;
  const int n0 = blockIdx.x * 64;

  // Weights: vectorized f32x4 -> f16x4 (global_load_b128 / ds_store_b64).
  for (int ch = tid; ch < (CCH * CCH) / 4; ch += 256) {
    float4 w = ((const float4*)Wconv)[ch];
    v4h h = {(_Float16)w.x, (_Float16)w.y, (_Float16)w.z, (_Float16)w.w};
    *(v4h*)(Ws + ch * 4) = h;
  }
  // X tile: coalesced f32x4 reads, transposed f16 stores.
  const float* xb = x + (size_t)b * CCH * HWPIX;
  for (int ch = tid; ch < (CCH * 64) / 4; ch += 256) {
    int c = ch >> 4, n4 = (ch & 15) * 4;
    float4 w = *(const float4*)(xb + (size_t)c * HWPIX + n0 + n4);
    XsT[(n4 + 0) * CCH + c] = (_Float16)w.x;
    XsT[(n4 + 1) * CCH + c] = (_Float16)w.y;
    XsT[(n4 + 2) * CCH + c] = (_Float16)w.z;
    XsT[(n4 + 3) * CCH + c] = (_Float16)w.w;
  }
  __syncthreads();

  const int wave = tid >> 5, lane = tid & 31;
  v8f acc[4] = {};
  for (int kb = 0; kb < CCH; kb += 32) {
    v16h af = load_A_lds(Ws, wave * 16 + (lane & 15), CCH, kb);
#pragma unroll
    for (int nt = 0; nt < 4; ++nt) {
      v16h bf = load_B_ldsT(XsT, nt * 16 + (lane & 15), CCH, kb);
      acc[nt] = wmma_f16(af, bf, acc[nt]);
    }
  }

#pragma unroll
  for (int nt = 0; nt < 4; ++nt) {
    unsigned n = n0 + nt * 16 + (lane & 15);
    unsigned hi = n / WW, wi = n % WW;
    unsigned jh = hi / HSUB, hrem = hi % HSUB;
    unsigned jw = wi / HSUB, wrem = wi % HSUB;
    unsigned j = jh * RR + jw;
    unsigned p = hrem * HSUB + wrem;
#pragma unroll
    for (int v = 0; v < 8; ++v) {
      int m = wave * 16 + v + ((lane >> 4) << 3);
      float val = acc[nt][v] + bconv[m];
      X2[((size_t)(b * CCH + m) * R2 + j) * NPOS + p] = (_Float16)val;
    }
  }
}

// ---------------------------------------------------------------------------
// Kernel 2: b2/c2 = W * U + bias over the 81-dim (padded to 96).
// grid (13, 512, 2), block 256 (8 waves). wave -> 16 positions, 6 Mtiles.
// U tile staged in LDS transposed [p][k] so B fragments are contiguous.
// ---------------------------------------------------------------------------
__global__ __launch_bounds__(256) void k2_proj(
    const _Float16* __restrict__ X2,
    const float* __restrict__ Wb, const float* __restrict__ bb,
    const float* __restrict__ Wc, const float* __restrict__ bcv,
    _Float16* __restrict__ b2, _Float16* __restrict__ c2) {
  __shared__ __align__(16) _Float16 Ws[R2P * R2P];    // [m][k], zero padded
  __shared__ __align__(16) _Float16 UsT[128 * R2P];   // [p][k], zero padded
  const int tid = threadIdx.x;
  const int bc = blockIdx.y;
  const int which = blockIdx.z;
  const float* W = which ? Wc : Wb;
  const float* bias = which ? bcv : bb;
  _Float16* out = which ? c2 : b2;
  const int p0b = blockIdx.x * 128;
  const _Float16* Xbc = X2 + (size_t)bc * R2 * NPOS;

  for (int i = tid; i < R2P * R2P; i += 256) {
    int m = i / R2P, k = i % R2P;
    Ws[i] = (m < R2 && k < R2) ? (_Float16)W[m * R2 + k] : (_Float16)0.0f;
  }
  // Coalesced along p from global; transposed store to LDS.
  for (int i = tid; i < R2P * 128; i += 256) {
    int k = i >> 7, p = i & 127;
    _Float16 v = (k < R2 && (p0b + p) < NPOS)
                     ? Xbc[(size_t)k * NPOS + p0b + p] : (_Float16)0.0f;
    UsT[p * R2P + k] = v;
  }
  __syncthreads();

  const int wave = tid >> 5, lane = tid & 31;
  const int pl = wave * 16 + (lane & 15);  // local position column
  v8f acc[6] = {};
  for (int kb = 0; kb < R2P; kb += 32) {
    v16h bf = load_B_ldsT(UsT, pl, R2P, kb);
#pragma unroll
    for (int mt = 0; mt < 6; ++mt) {
      v16h af = load_A_lds(Ws, mt * 16 + (lane & 15), R2P, kb);
      acc[mt] = wmma_f16(af, bf, acc[mt]);
    }
  }

  const int pcol = p0b + pl;
  if (pcol < NPOS) {
    _Float16* obc = out + (size_t)bc * R2 * NPOS;
#pragma unroll
    for (int mt = 0; mt < 6; ++mt)
#pragma unroll
      for (int v = 0; v < 8; ++v) {
        int m = mt * 16 + v + ((lane >> 4) << 3);
        if (m < R2)
          obc[(size_t)m * NPOS + pcol] = (_Float16)(acc[mt][v] + bias[m]);
      }
  }
}

// ---------------------------------------------------------------------------
// Kernel 3: K[bc] = b2 * c2^T * (1/9). grid (512), block 192 (6 waves).
// K-loop over 1600 positions with 96x32 LDS tiles (both K-contiguous).
// ---------------------------------------------------------------------------
__global__ __launch_bounds__(192) void k3_kmat(
    const _Float16* __restrict__ b2, const _Float16* __restrict__ c2,
    float* __restrict__ Km) {
  __shared__ __align__(16) _Float16 Bs[R2P * 32];  // A-tile: [m][k]
  __shared__ __align__(16) _Float16 Cs[R2P * 32];  // B-tile: [n][k] (natural)
  const int tid = threadIdx.x;
  const int bc = blockIdx.x;
  const _Float16* bb2 = b2 + (size_t)bc * R2 * NPOS;
  const _Float16* cc2 = c2 + (size_t)bc * R2 * NPOS;
  const int wave = tid >> 5, lane = tid & 31;
  const v8h zero8 = {};

  v8f acc[6] = {};
  for (int kb = 0; kb < NPOS; kb += 32) {
    __syncthreads();
    // 96*32/8 = 384 v8h chunks per array; 192 threads -> 2 chunks each.
#pragma unroll
    for (int t = 0; t < 2; ++t) {
      int vchunk = tid + t * 192;
      int m = vchunk >> 2, ko = (vchunk & 3) * 8;
      *(v8h*)(Bs + m * 32 + ko) =
          (m < R2) ? *(const v8h*)(bb2 + (size_t)m * NPOS + kb + ko) : zero8;
      *(v8h*)(Cs + m * 32 + ko) =
          (m < R2) ? *(const v8h*)(cc2 + (size_t)m * NPOS + kb + ko) : zero8;
    }
    __syncthreads();
    if (kb + 32 < NPOS) {  // pull next tiles toward L2 (global_prefetch_b8)
      __builtin_prefetch(&bb2[(size_t)(tid % R2) * NPOS + kb + 32], 0, 1);
      __builtin_prefetch(&cc2[(size_t)(tid % R2) * NPOS + kb + 32], 0, 1);
    }
    v16h af = load_A_lds(Bs, wave * 16 + (lane & 15), 32, 0);
#pragma unroll
    for (int nt = 0; nt < 6; ++nt) {
      v16h bf = load_B_ldsT(Cs, nt * 16 + (lane & 15), 32, 0);
      acc[nt] = wmma_f16(af, bf, acc[nt]);
    }
  }

  float* Kbc = Km + (size_t)bc * R2 * R2;
#pragma unroll
  for (int nt = 0; nt < 6; ++nt)
#pragma unroll
    for (int v = 0; v < 8; ++v) {
      int m = wave * 16 + v + ((lane >> 4) << 3);
      int n = nt * 16 + (lane & 15);
      if (m < R2 && n < R2) Kbc[m * R2 + n] = acc[nt][v] * (1.0f / 9.0f);
    }
}

// ---------------------------------------------------------------------------
// Kernel 4: G[bc] = Wa^T * K[bc] (81x81), g[bc] = K^T ba. grid(512), block 192.
// Folds the 'a' projection: y = u^T G + g, so a2 is never materialized.
// ---------------------------------------------------------------------------
__global__ __launch_bounds__(192) void k4_fold(
    const float* __restrict__ Wa, const float* __restrict__ ba,
    const float* __restrict__ Km, float* __restrict__ G,
    float* __restrict__ g) {
  __shared__ __align__(16) _Float16 As[R2P * R2P];   // As[j][i] = Wa[i][j]
  __shared__ __align__(16) _Float16 KsT[R2P * R2P];  // KsT[n][i] = K[i][n]
  const int tid = threadIdx.x;
  const int bc = blockIdx.x;
  const float* Kbc = Km + (size_t)bc * R2 * R2;

  for (int idx = tid; idx < R2P * R2P; idx += 192) {
    int r = idx / R2P, c = idx % R2P;
    As[idx] = (r < R2 && c < R2) ? (_Float16)Wa[c * R2 + r] : (_Float16)0.0f;
    KsT[idx] = (r < R2 && c < R2) ? (_Float16)Kbc[c * R2 + r] : (_Float16)0.0f;
  }
  __syncthreads();

  const int wave = tid >> 5, lane = tid & 31;
  v8f acc[6] = {};
  for (int kb = 0; kb < R2P; kb += 32) {
    v16h af = load_A_lds(As, wave * 16 + (lane & 15), R2P, kb);
#pragma unroll
    for (int nt = 0; nt < 6; ++nt) {
      v16h bf = load_B_ldsT(KsT, nt * 16 + (lane & 15), R2P, kb);
      acc[nt] = wmma_f16(af, bf, acc[nt]);
    }
  }

  float* Gbc = G + (size_t)bc * R2 * R2;
#pragma unroll
  for (int nt = 0; nt < 6; ++nt)
#pragma unroll
    for (int v = 0; v < 8; ++v) {
      int j = wave * 16 + v + ((lane >> 4) << 3);
      int n = nt * 16 + (lane & 15);
      if (j < R2 && n < R2) Gbc[j * R2 + n] = acc[nt][v];
    }

  if (tid < R2) {
    float s = 0.0f;
    for (int i = 0; i < R2; ++i) s += ba[i] * Kbc[i * R2 + tid];
    g[(size_t)bc * R2 + tid] = s;
  }
}

// ---------------------------------------------------------------------------
// Kernel 5: y^T[n][p] = sum_j G[j][n] U[j][p] + g[n]; scatter to NCHW f32 out.
// grid (50, 512), block 192 (6 waves). wave = n-tile, 2 position subtiles.
// U tile staged in LDS transposed [p][j].
// ---------------------------------------------------------------------------
__global__ __launch_bounds__(192) void k5_out(
    const _Float16* __restrict__ X2, const float* __restrict__ G,
    const float* __restrict__ g, float* __restrict__ out) {
  __shared__ __align__(16) _Float16 Gs[R2P * R2P];  // Gs[n][j] = G[j][n]
  __shared__ __align__(16) _Float16 UsT[32 * R2P];  // [p][j], zero padded
  const int tid = threadIdx.x;
  const int bc = blockIdx.y;
  const int p0 = blockIdx.x * 32;
  const float* Gbc = G + (size_t)bc * R2 * R2;
  const _Float16* Xbc = X2 + (size_t)bc * R2 * NPOS;

  for (int idx = tid; idx < R2P * R2P; idx += 192) {
    int r = idx / R2P, c = idx % R2P;  // r = n, c = j
    Gs[idx] = (r < R2 && c < R2) ? (_Float16)Gbc[c * R2 + r] : (_Float16)0.0f;
  }
  // Coalesced along p; transposed store.
  for (int i = tid; i < R2P * 32; i += 192) {
    int j = i >> 5, p = i & 31;
    UsT[p * R2P + j] =
        (j < R2) ? Xbc[(size_t)j * NPOS + p0 + p] : (_Float16)0.0f;
  }
  __syncthreads();

  const int wave = tid >> 5, lane = tid & 31;
  v8f acc[2] = {};
  for (int kb = 0; kb < R2P; kb += 32) {
    v16h af = load_A_lds(Gs, wave * 16 + (lane & 15), R2P, kb);
#pragma unroll
    for (int nt = 0; nt < 2; ++nt) {
      v16h bf = load_B_ldsT(UsT, nt * 16 + (lane & 15), R2P, kb);
      acc[nt] = wmma_f16(af, bf, acc[nt]);
    }
  }

#pragma unroll
  for (int nt = 0; nt < 2; ++nt)
#pragma unroll
    for (int v = 0; v < 8; ++v) {
      int n = wave * 16 + v + ((lane >> 4) << 3);
      if (n < R2) {
        int p = p0 + nt * 16 + (lane & 15);
        float val = acc[nt][v] + g[(size_t)bc * R2 + n];
        unsigned jh = (unsigned)n / RR, jw = (unsigned)n % RR;
        unsigned hrem = (unsigned)p / HSUB, wrem = (unsigned)p % HSUB;
        unsigned hi = jh * HSUB + hrem, wi = jw * HSUB + wrem;
        out[((size_t)bc * HH + hi) * WW + wi] = val;
      }
    }
}

// ---------------------------------------------------------------------------
extern "C" void kernel_launch(void* const* d_in, const int* in_sizes, int n_in,
                              void* d_out, int out_size, void* d_ws, size_t ws_size,
                              hipStream_t stream) {
  const float* x     = (const float*)d_in[0];
  const float* Wconv = (const float*)d_in[1];
  const float* bconv = (const float*)d_in[2];
  const float* Wa    = (const float*)d_in[3];
  const float* ba    = (const float*)d_in[4];
  const float* Wb    = (const float*)d_in[5];
  const float* bb    = (const float*)d_in[6];
  const float* Wc    = (const float*)d_in[7];
  const float* bcv   = (const float*)d_in[8];
  float* out = (float*)d_out;

  // Workspace layout (~425 MB): X2, b2, c2 in f16; K, G, g in f32.
  const size_t szX = (size_t)NBC * R2 * NPOS;   // 66.36M elems
  const size_t szK = (size_t)NBC * R2 * R2;     // 3.36M elems
  _Float16* X2 = (_Float16*)d_ws;
  _Float16* b2 = X2 + szX;
  _Float16* c2 = b2 + szX;
  float* Km = (float*)(c2 + szX);
  float* G  = Km + szK;
  float* g  = G + szK;

  k1_conv<<<dim3(HWPIX / 64, BATCH), 256, 0, stream>>>(x, Wconv, bconv, X2);
  k2_proj<<<dim3((NPOS + 127) / 128, NBC, 2), 256, 0, stream>>>(
      X2, Wb, bb, Wc, bcv, b2, c2);
  k3_kmat<<<dim3(NBC), 192, 0, stream>>>(b2, c2, Km);
  k4_fold<<<dim3(NBC), 192, 0, stream>>>(Wa, ba, Km, G, g);
  k5_out<<<dim3(NPOS / 32, NBC), 192, 0, stream>>>(X2, G, g, out);
}
